// SAGE_27814208209241
// MI455X (gfx1250) — matrix operand order repaired and checked by reading, not compiled, not run on hardware
//
#include <hip/hip_runtime.h>
#include <hip/hip_bf16.h>

typedef __attribute__((ext_vector_type(16))) __bf16 v16bf;
typedef __attribute__((ext_vector_type(8)))  float  v8f;

#define FLT_BIG 3.402823466e38f

// ---------------------------------------------------------------- utilities
__global__ void fill_kernel(float* __restrict__ p, long long n, float v) {
    long long i = (long long)blockIdx.x * blockDim.x + threadIdx.x;
    long long stride = (long long)gridDim.x * blockDim.x;
    for (; i < n; i += stride) p[i] = v;
}

// deg[dst] += 1 per edge (self-loops folded into dinv kernel as +1)
__global__ void degree_kernel(const int* __restrict__ dst, float* __restrict__ deg, long long E) {
    long long i = (long long)blockIdx.x * blockDim.x + threadIdx.x;
    long long stride = (long long)gridDim.x * blockDim.x;
    for (; i < E; i += stride) atomicAdd(&deg[dst[i]], 1.0f);
}

// dinv[i] = (deg[i]+1)^{-1/2}
__global__ void dinv_kernel(float* __restrict__ deg, int Nn) {
    int i = blockIdx.x * blockDim.x + threadIdx.x;
    if (i < Nn) deg[i] = rsqrtf(deg[i] + 1.0f);
}

// ---------------------------------------------------------------- WMMA GEMM
// C[M x NC] = A[M x K] @ B[K x NC] (row-major), bf16 operands, f32 accum.
// Compile-time K/NC; each wave owns a 16 x (NT*16) slab: one A fragment is
// reused for NT WMMAs per k-step; K-loop fully unrolled.
// MODE 0: plain store; MODE 1: store tanh(acc + bias[col]).
template<int K, int NC, int NT, int MODE>
__global__ void __launch_bounds__(256)
gemm_bf16_wmma(const float* __restrict__ A, const float* __restrict__ B,
               const float* __restrict__ bias, float* __restrict__ C, int M) {
    constexpr int NTILES = NC / 16;
    constexpr int WPR    = NTILES / NT;            // waves per 16-row panel
    int wid  = threadIdx.x >> 5;
    int lane = threadIdx.x & 31;
    int hl   = lane >> 4;                          // lane half (0/1)
    int lrow = lane & 15;

    long long gwave = (long long)blockIdx.x * 8 + wid;
    int mtile = (int)(gwave / WPR);
    int nt0   = (int)(gwave % WPR) * NT;           // first column tile
    if (mtile * 16 >= M) return;

    int arow = mtile * 16 + lrow;
    if (arow >= M) arow = M - 1;                   // clamp; store is guarded
    int bcol0 = nt0 * 16 + lrow;

    v8f acc[NT];
#pragma unroll
    for (int t = 0; t < NT; ++t) acc[t] = (v8f){};

#pragma unroll
    for (int kb = 0; kb < K; kb += 32) {
        v16bf a;
        const float* ap = A + (size_t)arow * K + kb + hl * 8;
#pragma unroll
        for (int i = 0; i < 8; ++i) {
            a[i]     = (__bf16)ap[i];              // K = kb + hl*8 + i
            a[8 + i] = (__bf16)ap[16 + i];         // K = kb + 16 + hl*8 + i
        }
#pragma unroll
        for (int t = 0; t < NT; ++t) {
            v16bf b;
            const float* bp = B + (size_t)(kb + hl * 8) * NC + bcol0 + t * 16;
#pragma unroll
            for (int i = 0; i < 16; ++i)
                b[i] = (__bf16)bp[(size_t)(i < 8 ? i : i + 8) * NC]; // K = kb + hl*8 + i (+8 for i>=8)
            acc[t] = __builtin_amdgcn_wmma_f32_16x16x32_bf16(
                false, a, false, b, (short)0, acc[t], false, false);
        }
    }

#pragma unroll
    for (int t = 0; t < NT; ++t) {
        int col = bcol0 + t * 16;
        float bv = (MODE == 1) ? bias[col] : 0.0f;
#pragma unroll
        for (int j = 0; j < 8; ++j) {
            int row = mtile * 16 + hl * 8 + j;
            if (row < M) {
                float v = acc[t][j];
                if (MODE == 1) v = tanhf(v + bv);
                C[(size_t)row * NC + col] = v;
            }
        }
    }
}

// ---------------------------------------------------------------- aggregation
// agg[dst] += h[src] * dinv[src]*dinv[dst] over E edges + N self-loops.
// One wave per edge; lane owns H/32 contiguous floats (b128/b64 gathers);
// prefetch the next grid-stride edge's source row to hide gather latency.
template<int H>
__global__ void __launch_bounds__(256)
aggregate_kernel(const float* __restrict__ h, const int* __restrict__ src,
                 const int* __restrict__ dst, const float* __restrict__ dinv,
                 float* __restrict__ agg, long long E, int Nn) {
    constexpr int V = H / 32;                      // floats per lane (4 or 2)
    int lane = threadIdx.x & 31;
    long long w = (long long)blockIdx.x * (blockDim.x >> 5) + (threadIdx.x >> 5);
    long long wstride = (long long)gridDim.x * (blockDim.x >> 5);
    long long total = E + Nn;
    for (long long e = w; e < total; e += wstride) {
        int s, d; float norm;
        if (e < E) { s = src[e]; d = dst[e]; norm = dinv[s] * dinv[d]; }
        else       { s = d = (int)(e - E); float di = dinv[s]; norm = di * di; }

        // prefetch next iteration's gather row (global_prefetch_b8)
        long long en = e + wstride;
        if (en < total) {
            int sn = (en < E) ? src[en] : (int)(en - E);
            __builtin_prefetch(h + (size_t)sn * H + lane * V, 0, 0);
        }

        const float* hp = h + (size_t)s * H + lane * V;
        float* ap = agg + (size_t)d * H + lane * V;
        float v[V];
        if (V == 4) {
            float4 t = *(const float4*)hp;
            v[0] = t.x; v[1] = t.y; v[V > 2 ? 2 : 0] = t.z; v[V > 3 ? 3 : 0] = t.w;
        } else {
            float2 t = *(const float2*)hp;
            v[0] = t.x; v[V > 1 ? 1 : 0] = t.y;
        }
#pragma unroll
        for (int i = 0; i < V; ++i)
            atomicAdd(ap + i, v[i] * norm);
    }
}

// x[r][c] = act(x[r][c] + b[c]); relu optional
__global__ void bias_act_kernel(float* __restrict__ x, const float* __restrict__ b,
                                long long M, int H, int do_relu) {
    long long i = (long long)blockIdx.x * blockDim.x + threadIdx.x;
    long long stride = (long long)gridDim.x * blockDim.x;
    long long n = M * H;
    for (; i < n; i += stride) {
        float v = x[i] + b[i % H];
        if (do_relu) v = fmaxf(v, 0.0f);
        x[i] = v;
    }
}

// logits[n][j] = a1[n][:] @ Wf2[:, j] + bf2[j]   (K=64, D2=8)
__global__ void logits_kernel(const float* __restrict__ a1, const float* __restrict__ Wf2,
                              const float* __restrict__ bf2, float* __restrict__ logits,
                              int Nn) {
    long long i = (long long)blockIdx.x * blockDim.x + threadIdx.x;
    long long n = (long long)Nn * 8;
    if (i >= n) return;
    int node = (int)(i >> 3), j = (int)(i & 7);
    const float* a = a1 + (size_t)node * 64;
    float s = bf2[j];
#pragma unroll
    for (int k = 0; k < 64; ++k) s += a[k] * Wf2[k * 8 + j];
    logits[i] = s;
}

// monotonic float <-> uint encoding for atomic max
__device__ __forceinline__ unsigned fenc(float f) {
    unsigned u = __float_as_uint(f);
    return (u & 0x80000000u) ? ~u : (u | 0x80000000u);
}
__device__ __forceinline__ float fdec(unsigned u) {
    return __uint_as_float((u & 0x80000000u) ? (u & 0x7FFFFFFFu) : ~u);
}

__global__ void __launch_bounds__(256)
colmax_kernel(const float* __restrict__ logits, unsigned* __restrict__ cmax, int Nn) {
    __shared__ float red[256 * 8];
    int tid = threadIdx.x;
    float lm[8];
#pragma unroll
    for (int j = 0; j < 8; ++j) lm[j] = -FLT_BIG;
    for (long long r = (long long)blockIdx.x * blockDim.x + tid; r < Nn;
         r += (long long)gridDim.x * blockDim.x) {
        const float* lp = logits + r * 8;
#pragma unroll
        for (int j = 0; j < 8; ++j) lm[j] = fmaxf(lm[j], lp[j]);
    }
#pragma unroll
    for (int j = 0; j < 8; ++j) red[tid * 8 + j] = lm[j];
    __syncthreads();
    for (int s = 128; s > 0; s >>= 1) {
        if (tid < s)
#pragma unroll
            for (int j = 0; j < 8; ++j)
                red[tid * 8 + j] = fmaxf(red[tid * 8 + j], red[(tid + s) * 8 + j]);
        __syncthreads();
    }
    if (tid < 8) atomicMax(&cmax[tid], fenc(red[tid]));
}

__global__ void __launch_bounds__(256)
colsum_kernel(const float* __restrict__ logits, const unsigned* __restrict__ cmax,
              float* __restrict__ csum, int Nn) {
    __shared__ float red[256 * 8];
    int tid = threadIdx.x;
    float cm[8], ls[8];
#pragma unroll
    for (int j = 0; j < 8; ++j) { cm[j] = fdec(cmax[j]); ls[j] = 0.0f; }
    for (long long r = (long long)blockIdx.x * blockDim.x + tid; r < Nn;
         r += (long long)gridDim.x * blockDim.x) {
        const float* lp = logits + r * 8;
#pragma unroll
        for (int j = 0; j < 8; ++j) ls[j] += expf(lp[j] - cm[j]);
    }
#pragma unroll
    for (int j = 0; j < 8; ++j) red[tid * 8 + j] = ls[j];
    __syncthreads();
    for (int s = 128; s > 0; s >>= 1) {
        if (tid < s)
#pragma unroll
            for (int j = 0; j < 8; ++j) red[tid * 8 + j] += red[(tid + s) * 8 + j];
        __syncthreads();
    }
    if (tid < 8) atomicAdd(&csum[tid], red[tid]);
}

__global__ void att_kernel(const float* __restrict__ logits, const unsigned* __restrict__ cmax,
                           const float* __restrict__ csum, float* __restrict__ att, int Nn) {
    long long i = (long long)blockIdx.x * blockDim.x + threadIdx.x;
    long long n = (long long)Nn * 8;
    if (i >= n) return;
    int j = (int)(i & 7);
    att[i] = expf(logits[i] - fdec(cmax[j])) / csum[j];
}

// ge[j*64+h] += sum_r att[r][j]*h2[r][h];  P[a*8+b] += sum_r att[r][a]*att[r][b]
__global__ void __launch_bounds__(512)
pool_kernel(const float* __restrict__ att, const float* __restrict__ h2,
            float* __restrict__ ge, float* __restrict__ P, int Nn) {
    int tid = threadIdx.x;
    int j = tid >> 6, hc = tid & 63;
    float acc = 0.0f;
    for (long long r = blockIdx.x; r < Nn; r += gridDim.x)
        acc += att[r * 8 + j] * h2[r * 64 + hc];
    atomicAdd(&ge[tid], acc);
    if (tid < 64) {
        int a = tid >> 3, b = tid & 7;
        float p = 0.0f;
        for (long long r = blockIdx.x; r < Nn; r += gridDim.x)
            p += att[r * 8 + a] * att[r * 8 + b];
        atomicAdd(&P[tid], p);
    }
}

// penalty + predictions + copy ge. out layout: [ge 512][penalty 1][pred 10]
__global__ void __launch_bounds__(512)
finalize_kernel(const float* __restrict__ ge, const float* __restrict__ P,
                const float* __restrict__ Wl, const float* __restrict__ bl,
                float* __restrict__ out) {
    __shared__ float rown[8];
    __shared__ float lg[10];
    int tid = threadIdx.x;
    out[tid] = ge[tid];
    if (tid < 8) {
        float s = 0.0f;
#pragma unroll
        for (int k = 0; k < 8; ++k) {
            float v = P[tid * 8 + k] - (tid == k ? 1.0f : 0.0f);
            s += v * v;
        }
        rown[tid] = sqrtf(s);
    }
    if (tid < 10) {
        float s = bl[tid];
        for (int i = 0; i < 512; ++i) s += ge[i] * Wl[i * 10 + tid];
        lg[tid] = s;
    }
    __syncthreads();
    if (tid == 0) {
        float pen = 0.0f;
#pragma unroll
        for (int j = 0; j < 8; ++j) pen += rown[j];
        out[512] = pen;
        float m = -FLT_BIG;
#pragma unroll
        for (int l = 0; l < 10; ++l) m = fmaxf(m, lg[l]);
        float se = 0.0f;
#pragma unroll
        for (int l = 0; l < 10; ++l) se += expf(lg[l] - m);
        float lse = m + logf(se);
#pragma unroll
        for (int l = 0; l < 10; ++l) out[513 + l] = lg[l] - lse;
    }
}

// ---------------------------------------------------------------- launch
extern "C" void kernel_launch(void* const* d_in, const int* in_sizes, int n_in,
                              void* d_out, int out_size, void* d_ws, size_t ws_size,
                              hipStream_t stream) {
    const int*   edges = (const int*)d_in[0];
    const float* X     = (const float*)d_in[1];
    const float* W1  = (const float*)d_in[2];
    const float* b1  = (const float*)d_in[3];
    const float* W2  = (const float*)d_in[4];
    const float* b2  = (const float*)d_in[5];
    const float* Wf1 = (const float*)d_in[6];
    const float* bf1 = (const float*)d_in[7];
    const float* Wf2 = (const float*)d_in[8];
    const float* bf2 = (const float*)d_in[9];
    const float* Wl  = (const float*)d_in[10];
    const float* bl  = (const float*)d_in[11];
    float* out = (float*)d_out;

    const int F = 128, H1 = 128, H2 = 64;
    long long E = in_sizes[0] / 2;
    int Nn = in_sizes[1] / F;
    const int* src = edges;
    const int* dst = edges + E;

    // workspace layout (floats)
    float* WS     = (float*)d_ws;
    float* p_dinv = WS;                                    // N
    float* p_A    = WS + Nn;                               // N*128
    float* p_B    = p_A + (size_t)Nn * 128;                // N*128
    float* p_sm   = p_B + (size_t)Nn * 128;                // small scratch
    unsigned* p_cmax = (unsigned*)p_sm;                    // 8
    float* p_csum = p_sm + 8;                              // 8
    float* p_ge   = p_sm + 16;                             // 512
    float* p_P    = p_sm + 528;                            // 64
    float* p_h2pre = p_A;                                  // N*64 (reuse)
    float* p_h2    = p_A + (size_t)Nn * 64;                // N*64
    float* p_a1    = p_B;                                  // N*64 (reuse)
    float* p_logits = p_B + (size_t)Nn * 64;               // N*8
    float* p_att    = p_logits + (size_t)Nn * 8;           // N*8

    int mtiles = (Nn + 15) / 16;

    // degrees -> dinv
    fill_kernel<<<1024, 256, 0, stream>>>(p_dinv, Nn, 0.0f);
    degree_kernel<<<4096, 256, 0, stream>>>(dst, p_dinv, E);
    dinv_kernel<<<(Nn + 255) / 256, 256, 0, stream>>>(p_dinv, Nn);

    // layer 1: XW1 -> aggregate -> +b1, relu
    {
        long long waves = (long long)mtiles * 2;           // WPR=2 for NC=128,NT=4
        gemm_bf16_wmma<128, 128, 4, 0><<<(int)((waves + 7) / 8), 256, 0, stream>>>(X, W1, nullptr, p_A, Nn);
    }
    fill_kernel<<<8192, 256, 0, stream>>>(p_B, (long long)Nn * H1, 0.0f);
    aggregate_kernel<128><<<32768, 256, 0, stream>>>(p_A, src, dst, p_dinv, p_B, E, Nn);
    bias_act_kernel<<<8192, 256, 0, stream>>>(p_B, b1, Nn, H1, 1);

    // layer 2: h1@W2 -> aggregate -> +b2
    {
        long long waves = (long long)mtiles * 1;           // WPR=1 for NC=64,NT=4
        gemm_bf16_wmma<128, 64, 4, 0><<<(int)((waves + 7) / 8), 256, 0, stream>>>(p_B, W2, nullptr, p_h2pre, Nn);
    }
    fill_kernel<<<8192, 256, 0, stream>>>(p_h2, (long long)Nn * H2, 0.0f);
    aggregate_kernel<64><<<32768, 256, 0, stream>>>(p_h2pre, src, dst, p_dinv, p_h2, E, Nn);
    bias_act_kernel<<<8192, 256, 0, stream>>>(p_h2, b2, Nn, H2, 0);

    // attention head: a1 = tanh(h2@Wf1 + bf1); logits = a1@Wf2 + bf2
    {
        long long waves = (long long)mtiles * 1;           // WPR=1 for NC=64,NT=4
        gemm_bf16_wmma<64, 64, 4, 1><<<(int)((waves + 7) / 8), 256, 0, stream>>>(p_h2, Wf1, bf1, p_a1, Nn);
    }
    logits_kernel<<<(int)(((long long)Nn * 8 + 255) / 256), 256, 0, stream>>>(p_a1, Wf2, bf2, p_logits, Nn);

    // column softmax over nodes
    fill_kernel<<<4, 256, 0, stream>>>(p_sm, 1024, 0.0f);  // cmax(enc 0 < enc(-inf)), csum, ge, P
    colmax_kernel<<<1024, 256, 0, stream>>>(p_logits, p_cmax, Nn);
    colsum_kernel<<<1024, 256, 0, stream>>>(p_logits, p_cmax, p_csum, Nn);
    att_kernel<<<(int)(((long long)Nn * 8 + 255) / 256), 256, 0, stream>>>(p_logits, p_cmax, p_csum, p_att, Nn);

    // pooling + finalize
    pool_kernel<<<2048, 512, 0, stream>>>(p_att, p_h2, p_ge, p_P, Nn);
    finalize_kernel<<<1, 512, 0, stream>>>(p_ge, p_P, Wl, bl, out);
}
